// S6InspiredHistoryCompressor_532575945372
// MI455X (gfx1250) — matrix operand order repaired and checked by reading, not compiled, not run on hardware
//
#include <hip/hip_runtime.h>
#include <hip/hip_bf16.h>
#include <math.h>

// ---------------- types for WMMA ----------------
typedef __attribute__((ext_vector_type(16))) __bf16 v16bf;
typedef __attribute__((ext_vector_type(8)))  __bf16 v8bf;
typedef __attribute__((ext_vector_type(8)))  float  v8f;

union Frag16 { v16bf v; v8bf h[2]; };

// native hardware convert (v_cvt_*bf16*), much cheaper than manual RNE bit-twiddling
__device__ __forceinline__ __bf16 f2bf(float f) { return (__bf16)f; }

__device__ __forceinline__ float siluf(float v) { return v / (1.0f + __expf(-v)); }
__device__ __forceinline__ float softplusf(float v) {
    return (v > 20.0f) ? v : log1pf(__expf(v));
}

// ---------------- problem constants ----------------
#define D_MODEL 1024
#define D_STATE 64
#define D_INNER 2048
#define DT_RANK 64
#define XP_DIM  192      // DT_RANK + 2*D_STATE
#define SEQ_LEN 2048
#define BATCH   4
#define MTOT    (SEQ_LEN * BATCH)   // 8192

// LDS row stride (bf16 elements): 144 bytes -> 16B aligned rows, bank-spread
#define LSTR 72

// ---------------- WMMA GEMM: out = A[M,K] @ Bw[N,K]^T + bias ----------------
// BM=128, BN=64, BK=64. 256 threads = 8 waves; each wave: 32x32 (2x2 WMMA tiles),
// two WMMA K-steps per staging round (8 v_wmma per round).
// mode 0: out0[row*ldc+col] = v + bias[col]
// mode 1: (W_in split) col<2048 -> out0[row*2048+col] = v+bias
//         col>=2048   -> out1[row*2048+col-2048] = silu(v+bias)/SEQ_LEN
// mode 2: out0[row*2048+col] = softplus(v + 2*bias[col])
__global__ __launch_bounds__(256) void gemm_wmma_bf16(
    const float* __restrict__ A, const float* __restrict__ Bw,
    const float* __restrict__ bias,
    float* __restrict__ out0, float* __restrict__ out1,
    int lda, int ldb, int ldc, int K, int mode)
{
    __shared__ __attribute__((aligned(16))) __bf16 As[128 * LSTR];
    __shared__ __attribute__((aligned(16))) __bf16 Bs[64 * LSTR];

    const int tid  = threadIdx.x;
    const int m0   = blockIdx.y * 128;
    const int n0   = blockIdx.x * 64;
    const int wave = tid >> 5;
    const int lane = tid & 31;
    const int lh   = lane >> 4;     // lane half
    const int l15  = lane & 15;
    const int wm   = wave >> 1;     // 0..3 (M subtile)
    const int wn   = wave & 1;      // 0..1 (N subtile)

    v8f acc[2][2] = {};

    const int arow = tid >> 1;            // 0..127
    const int acol = (tid & 1) * 32;      // 0 or 32
    const int brow = tid >> 2;            // 0..63
    const int bcol = (tid & 3) * 16;      // 0,16,32,48

    const float* srcA = A  + (size_t)(m0 + arow) * lda + acol;
    const float* srcB = Bw + (size_t)(n0 + brow) * ldb + bcol;

    for (int k0 = 0; k0 < K; k0 += 64) {
        __syncthreads();
        // stage A tile 128x64 (f32 -> bf16), 32 elements / thread
        {
            __bf16* dst = &As[arow * LSTR + acol];
            const float* s = srcA + k0;
            if (k0 + 64 < K) __builtin_prefetch(s + 64, 0, 1);
            #pragma unroll
            for (int i = 0; i < 32; i += 4) {
                float4 v = *(const float4*)(s + i);
                dst[i + 0] = f2bf(v.x); dst[i + 1] = f2bf(v.y);
                dst[i + 2] = f2bf(v.z); dst[i + 3] = f2bf(v.w);
            }
        }
        // stage B tile 64x64 (f32 -> bf16), 16 elements / thread
        {
            __bf16* dst = &Bs[brow * LSTR + bcol];
            const float* s = srcB + k0;
            if (k0 + 64 < K) __builtin_prefetch(s + 64, 0, 1);
            #pragma unroll
            for (int i = 0; i < 16; i += 4) {
                float4 v = *(const float4*)(s + i);
                dst[i + 0] = f2bf(v.x); dst[i + 1] = f2bf(v.y);
                dst[i + 2] = f2bf(v.z); dst[i + 3] = f2bf(v.w);
            }
        }
        __syncthreads();

        #pragma unroll
        for (int kk = 0; kk < 64; kk += 32) {
            Frag16 af[2], bf[2];
            #pragma unroll
            for (int i = 0; i < 2; i++) {
                const int row = wm * 32 + i * 16 + l15;
                af[i].h[0] = *(const v8bf*)&As[row * LSTR + kk + lh * 8];      // K = lh*8 + e
                af[i].h[1] = *(const v8bf*)&As[row * LSTR + kk + 16 + lh * 8]; // K = 16 + lh*8 + e
            }
            #pragma unroll
            for (int j = 0; j < 2; j++) {
                const int col = wn * 32 + j * 16 + l15;
                const v8bf* q = (const v8bf*)&Bs[col * LSTR + kk + lh * 8];  // K = lh*8 + e, 16 + lh*8 + e
                bf[j].h[0] = q[0];
                bf[j].h[1] = q[2];
            }
            #pragma unroll
            for (int i = 0; i < 2; i++)
                #pragma unroll
                for (int j = 0; j < 2; j++)
                    acc[i][j] = __builtin_amdgcn_wmma_f32_16x16x32_bf16(
                        false, af[i].v, false, bf[j].v, (short)0, acc[i][j], false, false);
        }
    }

    // epilogue
    #pragma unroll
    for (int i = 0; i < 2; i++) {
        #pragma unroll
        for (int j = 0; j < 2; j++) {
            const int col = n0 + wn * 32 + j * 16 + l15;
            const float bsc = (mode == 2) ? 2.0f * bias[col] : bias[col];
            #pragma unroll
            for (int r = 0; r < 8; r++) {
                const int row = m0 + wm * 32 + i * 16 + lh * 8 + r;
                float v = acc[i][j][r] + bsc;
                if (mode == 0) {
                    out0[(size_t)row * ldc + col] = v;
                } else if (mode == 1) {
                    if (col < D_INNER) out0[(size_t)row * D_INNER + col] = v;
                    else out1[(size_t)row * D_INNER + (col - D_INNER)] =
                             siluf(v) * (1.0f / (float)SEQ_LEN);
                } else {
                    out0[(size_t)row * D_INNER + col] = softplusf(v);
                }
            }
        }
    }
}

// ---------------- causal depthwise conv (K=4) + SiLU ----------------
__global__ __launch_bounds__(256) void conv_silu_kernel(
    const float* __restrict__ xin, const float* __restrict__ cw,
    const float* __restrict__ cb, float* __restrict__ xconv)
{
    size_t gid = (size_t)blockIdx.x * 256 + threadIdx.x;   // over MTOT*D_INNER
    int d = (int)(gid & (D_INNER - 1));
    int m = (int)(gid >> 11);
    int t = m >> 2, b = m & 3;
    float acc = cb[d];
    #pragma unroll
    for (int k = 0; k < 4; k++) {
        int tt = t - 3 + k;
        if (tt >= 0)
            acc += cw[d * 4 + k] * xin[(size_t)(tt * 4 + b) * D_INNER + d];
    }
    xconv[gid] = siluf(acc);
}

// ---------------- selective scan ----------------
// grid (256, 4): blockIdx.x = d-block (8 channels), blockIdx.y = batch.
// 8 waves/WG, one wave per channel d; lane owns n and n+32 state cells.
// y contribution accumulated with g = silu(z)/S folded in -> no per-step
// cross-lane reduction; single shuffle reduction at the end.
#define TCHUNK 64
__global__ __launch_bounds__(256) void scan_kernel(
    const float* __restrict__ xp,    const float* __restrict__ dtb,
    const float* __restrict__ xconv, const float* __restrict__ gbuf,
    const float* __restrict__ A_log, const float* __restrict__ Dp,
    const float* __restrict__ state_in,
    float* __restrict__ ymean, float* __restrict__ state_out)
{
    __shared__ float Bc[TCHUNK * 64];
    __shared__ float Cc[TCHUNK * 64];
    __shared__ float dtc[TCHUNK * 8];
    __shared__ float xc [TCHUNK * 8];
    __shared__ float gc [TCHUNK * 8];

    const int tid  = threadIdx.x;
    const int wave = tid >> 5;
    const int lane = tid & 31;
    const int b    = blockIdx.y;
    const int d    = blockIdx.x * 8 + wave;
    const int n0   = lane, n1 = lane + 32;

    const float A0 = -__expf(A_log[d * D_STATE + n0]);
    const float A1 = -__expf(A_log[d * D_STATE + n1]);
    float s0 = state_in[((size_t)b * D_INNER + d) * D_STATE + n0];
    float s1 = state_in[((size_t)b * D_INNER + d) * D_STATE + n1];
    float acc0 = 0.f, acc1 = 0.f, xg = 0.f;

    for (int t0 = 0; t0 < SEQ_LEN; t0 += TCHUNK) {
        __syncthreads();
        // stage B_seq / C_seq  (64 t x 64 n)
        for (int idx = tid; idx < TCHUNK * 64; idx += 256) {
            int tt = idx >> 6, nn = idx & 63;
            size_t m = (size_t)(t0 + tt) * 4 + b;
            Bc[idx] = xp[m * XP_DIM + DT_RANK + nn];
            Cc[idx] = xp[m * XP_DIM + DT_RANK + D_STATE + nn];
        }
        // stage dt / x / g  (64 t x 8 d)
        for (int idx = tid; idx < TCHUNK * 8; idx += 256) {
            int tt = idx >> 3, dd = idx & 7;
            size_t m = (size_t)(t0 + tt) * 4 + b;
            int dg = blockIdx.x * 8 + dd;
            dtc[idx] = dtb  [m * D_INNER + dg];
            xc [idx] = xconv[m * D_INNER + dg];
            gc [idx] = gbuf [m * D_INNER + dg];
        }
        __syncthreads();
        #pragma unroll 4
        for (int tt = 0; tt < TCHUNK; tt++) {
            float dtv = dtc[tt * 8 + wave];
            float xv  = xc [tt * 8 + wave];
            float gv  = gc [tt * 8 + wave];
            float bt0 = Bc[tt * 64 + n0], bt1 = Bc[tt * 64 + n1];
            float ct0 = Cc[tt * 64 + n0], ct1 = Cc[tt * 64 + n1];
            float dxv = dtv * xv;
            s0 = s0 * __expf(dtv * A0) + dxv * bt0;
            s1 = s1 * __expf(dtv * A1) + dxv * bt1;
            acc0 += gv * s0 * ct0;
            acc1 += gv * s1 * ct1;
            xg   += gv * xv;
        }
    }

    state_out[((size_t)b * D_INNER + d) * D_STATE + n0] = s0;
    state_out[((size_t)b * D_INNER + d) * D_STATE + n1] = s1;

    float r = acc0 + acc1;
    #pragma unroll
    for (int off = 16; off; off >>= 1) r += __shfl_xor(r, off, 32);
    if (lane == 0) ymean[(size_t)b * D_INNER + d] = r + Dp[d] * xg;
}

// ---------------- out projection: outpre[b,n] = ymean[b,:] . W_out[n,:] + b_out ----
__global__ __launch_bounds__(256) void outproj_kernel(
    const float* __restrict__ ymean, const float* __restrict__ W_out,
    const float* __restrict__ b_out, float* __restrict__ outpre)
{
    int wave = threadIdx.x >> 5, lane = threadIdx.x & 31;
    int idx = blockIdx.x * 8 + wave;         // 0..4095
    int b = idx >> 10, n = idx & 1023;
    const float* yr = ymean + (size_t)b * D_INNER;
    const float* wr = W_out + (size_t)n * D_INNER;
    float s = 0.f;
    for (int k = lane; k < D_INNER; k += 32) s += yr[k] * wr[k];
    #pragma unroll
    for (int off = 16; off; off >>= 1) s += __shfl_xor(s, off, 32);
    if (lane == 0) outpre[idx] = s + b_out[n];
}

// ---------------- layernorm over last dim (1024) ----------------
__global__ __launch_bounds__(256) void lnorm_kernel(
    const float* __restrict__ pre, const float* __restrict__ gam,
    const float* __restrict__ bet, float* __restrict__ out)
{
    __shared__ float sA[8], sB[8];
    int b = blockIdx.x, tid = threadIdx.x;
    float v[4]; float s = 0.f, s2 = 0.f;
    #pragma unroll
    for (int i = 0; i < 4; i++) {
        v[i] = pre[b * 1024 + tid * 4 + i];
        s += v[i]; s2 += v[i] * v[i];
    }
    #pragma unroll
    for (int off = 16; off; off >>= 1) {
        s  += __shfl_xor(s,  off, 32);
        s2 += __shfl_xor(s2, off, 32);
    }
    int wave = tid >> 5, lane = tid & 31;
    if (lane == 0) { sA[wave] = s; sB[wave] = s2; }
    __syncthreads();
    if (tid == 0) {
        float a = 0.f, c = 0.f;
        for (int i = 0; i < 8; i++) { a += sA[i]; c += sB[i]; }
        sA[0] = a; sB[0] = c;
    }
    __syncthreads();
    float mean = sA[0] * (1.0f / 1024.0f);
    float var  = sB[0] * (1.0f / 1024.0f) - mean * mean;
    float rstd = rsqrtf(var + 1e-5f);
    #pragma unroll
    for (int i = 0; i < 4; i++) {
        int n = tid * 4 + i;
        out[b * 1024 + n] = (v[i] - mean) * rstd * gam[n] + bet[n];
    }
}

// ---------------- launch ----------------
extern "C" void kernel_launch(void* const* d_in, const int* in_sizes, int n_in,
                              void* d_out, int out_size, void* d_ws, size_t ws_size,
                              hipStream_t stream) {
    const float* x      = (const float*)d_in[0];
    const float* state  = (const float*)d_in[1];
    const float* W_in   = (const float*)d_in[2];
    const float* b_in   = (const float*)d_in[3];
    const float* conv_w = (const float*)d_in[4];
    const float* conv_b = (const float*)d_in[5];
    const float* W_xp   = (const float*)d_in[6];
    const float* b_xp   = (const float*)d_in[7];
    const float* W_dt   = (const float*)d_in[8];
    const float* b_dt   = (const float*)d_in[9];
    const float* A_log  = (const float*)d_in[10];
    const float* Dp     = (const float*)d_in[11];
    const float* W_out  = (const float*)d_in[12];
    const float* b_out  = (const float*)d_in[13];
    const float* ln_g   = (const float*)d_in[14];
    const float* ln_b   = (const float*)d_in[15];
    float* out = (float*)d_out;                 // [0,4096): out ; [4096,...): final_state
    float* ws  = (float*)d_ws;

    const size_t nMD   = (size_t)MTOT * D_INNER;        // 16,777,216
    float* xinner = ws;                                  // [MTOT, D_INNER]
    float* gbuf   = xinner + nMD;                        // silu(z)/S
    float* xconv  = gbuf + nMD;                          // [MTOT, D_INNER]
    float* xp     = xconv + nMD;                         // [MTOT, 192]
    float* dtb    = xp + (size_t)MTOT * XP_DIM;          // [MTOT, D_INNER]
    float* ymean  = dtb + nMD;                           // [BATCH, D_INNER]
    float* outpre = ymean + (size_t)BATCH * D_INNER;     // [BATCH, D_MODEL]

    // GEMM 1: xz = x @ W_in^T  (M=8192, N=4096, K=1024), split + silu(z)/S epilogue
    gemm_wmma_bf16<<<dim3(2 * D_INNER / 64, MTOT / 128), 256, 0, stream>>>(
        x, W_in, b_in, xinner, gbuf, D_MODEL, D_MODEL, 0, D_MODEL, 1);

    // causal depthwise conv + SiLU
    conv_silu_kernel<<<(unsigned)(nMD / 256), 256, 0, stream>>>(
        xinner, conv_w, conv_b, xconv);

    // GEMM 2: xp = x_conv @ W_xp^T  (M=8192, N=192, K=2048)
    gemm_wmma_bf16<<<dim3(XP_DIM / 64, MTOT / 128), 256, 0, stream>>>(
        xconv, W_xp, b_xp, xp, nullptr, D_INNER, D_INNER, XP_DIM, D_INNER, 0);

    // GEMM 3: dt = softplus(xp[:, :64] @ W_dt^T + 2*b_dt)  (M=8192, N=2048, K=64)
    gemm_wmma_bf16<<<dim3(D_INNER / 64, MTOT / 128), 256, 0, stream>>>(
        xp, W_dt, b_dt, dtb, nullptr, XP_DIM, DT_RANK, 0, DT_RANK, 2);

    // selective scan + fused silu(z)*mean-over-S accumulation + final state
    scan_kernel<<<dim3(D_INNER / 8, BATCH), 256, 0, stream>>>(
        xp, dtb, xconv, gbuf, A_log, Dp, state, ymean, out + BATCH * D_MODEL);

    // out projection
    outproj_kernel<<<512, 256, 0, stream>>>(ymean, W_out, b_out, outpre);

    // layernorm -> d_out[0:4096]
    lnorm_kernel<<<BATCH, 256, 0, stream>>>(outpre, ln_g, ln_b, out);
}
